// Res_up_8658654069156
// MI455X (gfx1250) — matrix operand order repaired and checked by reading, not compiled, not run on hardware
//
#include <hip/hip_runtime.h>
#include <math.h>

// ============================================================================
// Res_up block for MI455X (gfx1250, wave32, WMMA).
//
// Strategy (reasoned from MI455X specs, no runtime available):
//  * ~115 GFLOP of small GEMMs (edge/node MLPs) vs ~2-3 GB of gather traffic
//    => balanced; use v_wmma_f32_16x16x32_bf16 (bf16 in, f32 accumulate).
//  * Weights converted once per launch to bf16 and pre-TRANSPOSED to
//    [dout][din] so each lane's B-fragment is 16 contiguous K values
//    (k = 16h + s, same order as the A fragment). Weights stay L2-resident (192 MB L2).
//  * Unpool handled by inverse-index maps (inv_node over N_F, inv_edge over
//    E_F); rows not present map to zero input vectors, which still flow
//    through the MLP (bias + LayerNorm make MLP(0) != 0, as in reference).
//  * segment_sum fused into the edge-MLP epilogue as f32 global atomics.
//  * One generic fused kernel: gather-concat -> WMMA GEMM1 + ReLU ->
//    WMMA GEMM2 -> LayerNorm -> {bf16 store | atomic agg | residual+LeakyReLU}.
//
// Input order assumptions: top-level dict insertion order; nested param dicts
// in jax tree_leaves (sorted-key) order: W1, W2, b, b1, b2, g; "edge"<"node".
// Index arrays cast to const int* per the harness contract.
// ============================================================================

typedef __attribute__((ext_vector_type(16))) __bf16 bf16x16;
typedef __attribute__((ext_vector_type(8)))  float  f32x8;

struct Part {
    const void* src;      // feature source (f32 or bf16-as-ushort), row-major
    int width;            // feature width of this part
    int is_bf16;          // 1 if src is bf16 (ushort), else f32
    const int* rowidx;    // per-row index (e.g. src/dst of edge); null = identity
    const int* inv;       // secondary inverse map (unpool); null = none; -1 = zero row
};

struct MlpArgs {
    int rows, din, dout;
    Part pt0, pt1, pt2;                 // concatenated input parts (sum widths == din)
    const unsigned short* W1T;          // bf16 [dout][din]
    const float* bias1;
    const unsigned short* W2T;          // bf16 [dout][dout]
    const float* bias2;
    const float* lng;                   // LayerNorm gamma
    const float* lnb;                   // LayerNorm beta
    unsigned short* out_bf;             // optional bf16 output [rows][dout]
    float* out_f32;                     // optional f32 output  [rows][dout]
    const float* resid;                 // optional residual; if set: LeakyReLU(y+resid)
    float* agg;                         // optional f32 segment-sum target
    const int* agg_idx;                 // dst index per row for agg
};

__device__ __forceinline__ unsigned short f32_to_bf16(float f) {
    unsigned int u = __float_as_uint(f);
    unsigned int r = u + 0x7FFFu + ((u >> 16) & 1u);   // round-to-nearest-even
    return (unsigned short)(r >> 16);
}

__device__ __forceinline__ float bf16_to_f32(unsigned short h) {
    return __uint_as_float(((unsigned int)h) << 16);
}

// ---------------------------------------------------------------------------
// Fused 2-layer MLP + LayerNorm, 16 rows per block, dout/16 waves per block.
// ---------------------------------------------------------------------------
__global__ void fused_mlp(MlpArgs a) {
    __shared__ unsigned short Abuf[16 * 384];   // gathered bf16 input tile
    __shared__ unsigned short Hbuf[16 * 128];   // hidden (post-ReLU) bf16
    __shared__ float          Sbuf[16 * 128];   // GEMM2 output (pre-LN) f32
    __shared__ float          red0[256], red1[256];
    __shared__ float          rmu[16], rrs[16];

    const int tid  = threadIdx.x;
    const int lane = tid & 31;
    const int wave = tid >> 5;
    const int row0 = blockIdx.x * 16;
    const int din  = a.din;
    const int dout = a.dout;

    // ---- Stage 1: gather + concat + convert into LDS (bf16) ----
    for (int idx = tid; idx < 16 * din; idx += blockDim.x) {
        const int r = idx / din;
        const int c = idx - r * din;
        const int grow = row0 + r;
        unsigned short hv = 0;
        if (grow < a.rows) {
            Part P; int cc = c;
            if (cc < a.pt0.width) { P = a.pt0; }
            else { cc -= a.pt0.width;
                   if (cc < a.pt1.width) { P = a.pt1; }
                   else { cc -= a.pt1.width; P = a.pt2; } }
            int j = P.rowidx ? P.rowidx[grow] : grow;
            if (P.inv) j = (j >= 0) ? P.inv[j] : -1;
            if (j >= 0) {
                if (P.is_bf16)
                    hv = ((const unsigned short*)P.src)[(long)j * P.width + cc];
                else
                    hv = f32_to_bf16(((const float*)P.src)[(long)j * P.width + cc]);
            }
        }
        Abuf[idx] = hv;
    }
    __syncthreads();

    // Fragment geometry (wave32; k = 16h + s in both A and B):
    const int m    = lane & 15;              // A row for this lane
    const int kb_a = (lane >> 4) << 4;       // A K-base: 0 or 16
    const int kb_b = (lane >> 4) << 4;       // B K-base: 0 or 16
    const int ncol = wave * 16 + (lane & 15);// B column (== output column)

    union Frag { unsigned short u[16]; bf16x16 v; };

    // ---- Stage 2: GEMM1 (din -> dout), ReLU ----
    f32x8 acc = {};
    for (int k0 = 0; k0 < din; k0 += 32) {
        Frag fa, fb;
        const unsigned short* ap = &Abuf[m * din + k0 + kb_a];
#pragma unroll
        for (int e = 0; e < 16; ++e) fa.u[e] = ap[e];
        const unsigned short* bp = &a.W1T[(long)ncol * din + k0 + kb_b];
        __builtin_prefetch(bp + 32, 0, 1);
#pragma unroll
        for (int e = 0; e < 16; ++e) fb.u[e] = bp[e];
        acc = __builtin_amdgcn_wmma_f32_16x16x32_bf16(false, fa.v, false, fb.v,
                                                      (short)0, acc, false, false);
    }
    {
        const float bv = a.bias1[ncol];
#pragma unroll
        for (int i = 0; i < 8; ++i) {
            const int mm = i + ((lane >> 4) << 3);
            float h = acc[i] + bv;
            h = h > 0.f ? h : 0.f;
            Hbuf[mm * dout + ncol] = f32_to_bf16(h);
        }
    }
    __syncthreads();

    // ---- Stage 3: GEMM2 (dout -> dout) ----
    f32x8 acc2 = {};
    for (int k0 = 0; k0 < dout; k0 += 32) {
        Frag fa, fb;
        const unsigned short* ap = &Hbuf[m * dout + k0 + kb_a];
#pragma unroll
        for (int e = 0; e < 16; ++e) fa.u[e] = ap[e];
        const unsigned short* bp = &a.W2T[(long)ncol * dout + k0 + kb_b];
#pragma unroll
        for (int e = 0; e < 16; ++e) fb.u[e] = bp[e];
        acc2 = __builtin_amdgcn_wmma_f32_16x16x32_bf16(false, fa.v, false, fb.v,
                                                       (short)0, acc2, false, false);
    }
    {
        const float bv = a.bias2[ncol];
#pragma unroll
        for (int i = 0; i < 8; ++i) {
            const int mm = i + ((lane >> 4) << 3);
            Sbuf[mm * dout + ncol] = acc2[i] + bv;
        }
    }
    __syncthreads();

    // ---- Stage 4: LayerNorm stats (per row over dout) ----
    {
        const int r = tid & 15;
        const int grp = tid >> 4;
        const int G = blockDim.x >> 4;
        float s1 = 0.f, s2 = 0.f;
        for (int c = grp; c < dout; c += G) {
            const float v = Sbuf[r * dout + c];
            s1 += v; s2 += v * v;
        }
        red0[grp * 16 + r] = s1;
        red1[grp * 16 + r] = s2;
        __syncthreads();
        if (grp == 0) {
            float t1 = 0.f, t2 = 0.f;
            for (int k = 0; k < G; ++k) { t1 += red0[k * 16 + r]; t2 += red1[k * 16 + r]; }
            const float mu = t1 / (float)dout;
            const float var = t2 / (float)dout - mu * mu;
            rmu[r] = mu;
            rrs[r] = rsqrtf(var + 1e-5f);
        }
        __syncthreads();
    }

    // ---- Stage 5: scale/shift + epilogue (store / atomic agg / residual) ----
    for (int idx = tid; idx < 16 * dout; idx += blockDim.x) {
        const int r = idx / dout;
        const int c = idx - r * dout;
        const int grow = row0 + r;
        if (grow >= a.rows) continue;
        const float y = (Sbuf[idx] - rmu[r]) * rrs[r] * a.lng[c] + a.lnb[c];
        const long o = (long)grow * dout + c;
        if (a.out_bf) a.out_bf[o] = f32_to_bf16(y);
        if (a.agg) {
            const int d = a.agg_idx[grow];
            atomicAdd(&a.agg[(long)d * dout + c], y);
        }
        if (a.out_f32) {
            float v = y;
            if (a.resid) {
                v += a.resid[o];
                v = v > 0.f ? v : 0.01f * v;   // LeakyReLU merge
            }
            a.out_f32[o] = v;
        }
    }
}

// ---------------------------------------------------------------------------
// Prep kernels
// ---------------------------------------------------------------------------
__global__ void k_transpose_bf16(const float* __restrict__ W,
                                 unsigned short* __restrict__ out,
                                 int din, int dout) {
    const int i = blockIdx.x * blockDim.x + threadIdx.x;
    if (i < din * dout) {
        const int k = i / dout, n = i - k * dout;
        out[(long)n * din + k] = f32_to_bf16(W[i]);
    }
}

__global__ void k_fill_i32(int* p, int n, int v) {
    const int i = blockIdx.x * blockDim.x + threadIdx.x;
    if (i < n) p[i] = v;
}

__global__ void k_zero_f32(float* p, long n) {
    const long i = (long)blockIdx.x * blockDim.x + threadIdx.x;
    if (i < n) p[i] = 0.f;
}

__global__ void k_scatter_inv(const int* __restrict__ idx, int* __restrict__ inv, int n) {
    const int i = blockIdx.x * blockDim.x + threadIdx.x;
    if (i < n) inv[idx[i]] = i;
}

// ---------------------------------------------------------------------------
// Host launch
// ---------------------------------------------------------------------------
extern "C" void kernel_launch(void* const* d_in, const int* in_sizes, int n_in,
                              void* d_out, int out_size, void* d_ws, size_t ws_size,
                              hipStream_t stream) {
    const float* x      = (const float*)d_in[0];
    const float* eattr  = (const float*)d_in[1];
    const int*   eic    = (const int*)d_in[2];
    const int*   eif    = (const int*)d_in[3];
    const int*   mask   = (const int*)d_in[4];
    const int*   eidx   = (const int*)d_in[5];

    const int CIN = 128;
    const int NC = in_sizes[0] / CIN;     // 32768
    const int EC = in_sizes[1] / CIN;     // 200000
    const int EF = in_sizes[3] / 2;       // 400000
    const int NF = 65536;                 // n_up (device scalar; reference N_F)

    const int* src_c = eic;        const int* dst_c = eic + EC;
    const int* src_f = eif;        const int* dst_f = eif + EF;

    struct MlpP { const float *W1, *W2, *lnb, *b1, *b2, *g; int din, dout; };
    auto getmlp = [&](int base, int din, int dout) {
        MlpP m;
        m.W1  = (const float*)d_in[base + 0];
        m.W2  = (const float*)d_in[base + 1];
        m.lnb = (const float*)d_in[base + 2];   // "b"
        m.b1  = (const float*)d_in[base + 3];
        m.b2  = (const float*)d_in[base + 4];
        m.g   = (const float*)d_in[base + 5];
        m.din = din; m.dout = dout;
        return m;
    };
    MlpP p1e = getmlp(7, 384, 64),  p1n = getmlp(13, 192, 64);
    MlpP p2e = getmlp(19, 192, 128), p2n = getmlp(25, 192, 128);
    MlpP pse = getmlp(31, 384, 128), psn = getmlp(37, 256, 128);

    // ---- workspace carving ----
    size_t off = 0;
    auto carve = [&](size_t bytes) -> void* {
        void* p = (char*)d_ws + off;
        off += (bytes + 255) & ~(size_t)255;
        return p;
    };
    auto prepw = [&](const MlpP& m, unsigned short*& w1t, unsigned short*& w2t) {
        w1t = (unsigned short*)carve((size_t)m.din * m.dout * 2);
        w2t = (unsigned short*)carve((size_t)m.dout * m.dout * 2);
        int n1 = m.din * m.dout, n2 = m.dout * m.dout;
        k_transpose_bf16<<<(n1 + 255) / 256, 256, 0, stream>>>(m.W1, w1t, m.din, m.dout);
        k_transpose_bf16<<<(n2 + 255) / 256, 256, 0, stream>>>(m.W2, w2t, m.dout, m.dout);
    };
    unsigned short *p1e_w1, *p1e_w2, *p1n_w1, *p1n_w2;
    unsigned short *p2e_w1, *p2e_w2, *p2n_w1, *p2n_w2;
    unsigned short *pse_w1, *pse_w2, *psn_w1, *psn_w2;
    prepw(p1e, p1e_w1, p1e_w2);  prepw(p1n, p1n_w1, p1n_w2);
    prepw(p2e, p2e_w1, p2e_w2);  prepw(p2n, p2n_w1, p2n_w2);
    prepw(pse, pse_w1, pse_w2);  prepw(psn, psn_w1, psn_w2);

    int*   inv_node = (int*)carve((size_t)NF * 4);
    int*   inv_edge = (int*)carve((size_t)EF * 4);
    float* agg_s = (float*)carve((size_t)NF * 128 * 4);
    float* agg1  = (float*)carve((size_t)NC * 64 * 4);
    float* agg2  = (float*)carve((size_t)NF * 128 * 4);
    float* xskip = (float*)carve((size_t)NF * 128 * 4);
    unsigned short* e1 = (unsigned short*)carve((size_t)EC * 64 * 2);
    unsigned short* x1 = (unsigned short*)carve((size_t)NC * 64 * 2);
    (void)ws_size; (void)n_in; (void)out_size;

    // inverse maps + zeroed aggregation buffers (every call; atomics accumulate)
    k_fill_i32<<<(NF + 255) / 256, 256, 0, stream>>>(inv_node, NF, -1);
    k_fill_i32<<<(EF + 255) / 256, 256, 0, stream>>>(inv_edge, EF, -1);
    k_zero_f32<<<(int)(((long)NF * 128 + 255) / 256), 256, 0, stream>>>(agg_s, (long)NF * 128);
    k_zero_f32<<<(int)(((long)NC * 64  + 255) / 256), 256, 0, stream>>>(agg1,  (long)NC * 64);
    k_zero_f32<<<(int)(((long)NF * 128 + 255) / 256), 256, 0, stream>>>(agg2,  (long)NF * 128);
    k_scatter_inv<<<(NC + 255) / 256, 256, 0, stream>>>(mask, inv_node, NC);
    k_scatter_inv<<<(EC + 255) / 256, 256, 0, stream>>>(eidx, inv_edge, EC);

    auto mkpart = [](const void* s, int w, int bf, const int* ri, const int* inv) {
        Part p; p.src = s; p.width = w; p.is_bf16 = bf; p.rowidx = ri; p.inv = inv;
        return p;
    };
    auto launch = [&](MlpArgs& A) {
        const int waves = A.dout / 16;
        fused_mlp<<<(A.rows + 15) / 16, waves * 32, 0, stream>>>(A);
    };
    Part none = mkpart(nullptr, 0, 0, nullptr, nullptr);

    // ---- skip branch: ps.edge over fine graph (zero-padded via inv maps) ----
    {
        MlpArgs A = {};
        A.rows = EF; A.din = 384; A.dout = 128;
        A.pt0 = mkpart(x, 128, 0, src_f, inv_node);
        A.pt1 = mkpart(x, 128, 0, dst_f, inv_node);
        A.pt2 = mkpart(eattr, 128, 0, nullptr, inv_edge);
        A.W1T = pse_w1; A.bias1 = pse.b1; A.W2T = pse_w2; A.bias2 = pse.b2;
        A.lng = pse.g;  A.lnb = pse.lnb;
        A.agg = agg_s;  A.agg_idx = dst_f;
        launch(A);
    }
    // ---- ps.node -> x_skip ----
    {
        MlpArgs A = {};
        A.rows = NF; A.din = 256; A.dout = 128;
        A.pt0 = mkpart(x, 128, 0, nullptr, inv_node);
        A.pt1 = mkpart(agg_s, 128, 0, nullptr, nullptr);
        A.pt2 = none;
        A.W1T = psn_w1; A.bias1 = psn.b1; A.W2T = psn_w2; A.bias2 = psn.b2;
        A.lng = psn.g;  A.lnb = psn.lnb;
        A.out_f32 = xskip;
        launch(A);
    }
    // ---- main branch: p1.edge on coarse graph -> e1 (+agg1) ----
    {
        MlpArgs A = {};
        A.rows = EC; A.din = 384; A.dout = 64;
        A.pt0 = mkpart(x, 128, 0, src_c, nullptr);
        A.pt1 = mkpart(x, 128, 0, dst_c, nullptr);
        A.pt2 = mkpart(eattr, 128, 0, nullptr, nullptr);
        A.W1T = p1e_w1; A.bias1 = p1e.b1; A.W2T = p1e_w2; A.bias2 = p1e.b2;
        A.lng = p1e.g;  A.lnb = p1e.lnb;
        A.out_bf = e1; A.agg = agg1; A.agg_idx = dst_c;
        launch(A);
    }
    // ---- p1.node -> x1 ----
    {
        MlpArgs A = {};
        A.rows = NC; A.din = 192; A.dout = 64;
        A.pt0 = mkpart(x, 128, 0, nullptr, nullptr);
        A.pt1 = mkpart(agg1, 64, 0, nullptr, nullptr);
        A.pt2 = none;
        A.W1T = p1n_w1; A.bias1 = p1n.b1; A.W2T = p1n_w2; A.bias2 = p1n.b2;
        A.lng = p1n.g;  A.lnb = p1n.lnb;
        A.out_bf = x1;
        launch(A);
    }
    // ---- p2.edge over fine graph (x1/e1 unpooled via inv maps) -> agg2 ----
    {
        MlpArgs A = {};
        A.rows = EF; A.din = 192; A.dout = 128;
        A.pt0 = mkpart(x1, 64, 1, src_f, inv_node);
        A.pt1 = mkpart(x1, 64, 1, dst_f, inv_node);
        A.pt2 = mkpart(e1, 64, 1, nullptr, inv_edge);
        A.W1T = p2e_w1; A.bias1 = p2e.b1; A.W2T = p2e_w2; A.bias2 = p2e.b2;
        A.lng = p2e.g;  A.lnb = p2e.lnb;
        A.agg = agg2; A.agg_idx = dst_f;
        launch(A);
    }
    // ---- p2.node + residual(x_skip) + LeakyReLU -> d_out ----
    {
        MlpArgs A = {};
        A.rows = NF; A.din = 192; A.dout = 128;
        A.pt0 = mkpart(x1, 64, 1, nullptr, inv_node);
        A.pt1 = mkpart(agg2, 128, 0, nullptr, nullptr);
        A.pt2 = none;
        A.W1T = p2n_w1; A.bias1 = p2n.b1; A.W2T = p2n_w2; A.bias2 = p2n.b2;
        A.lng = p2n.g;  A.lnb = p2n.lnb;
        A.out_f32 = (float*)d_out; A.resid = xskip;
        launch(A);
    }
}
